// ChainCRF_63831803953589
// MI455X (gfx1250) — compile-verified, matrix operand order checked by reading, not run on hardware
//
#include <hip/hip_runtime.h>
#include <hip/hip_bf16.h>

// Problem constants (B, T, D, NUM_LABELS) = (32, 256, 512, 64); L = 65
#define BB     32
#define TT     256
#define DD     512
#define LLAB   65            // NUM_LABELS + 1
#define NTRANS (LLAB * LLAB) // 4225 transition rows
#define NTOT   (NTRANS + LLAB) // 4290 = trans + state rows
#define NPAD   4352          // 34 * 128, padded N for the fused GEMM

typedef __attribute__((ext_vector_type(16))) __bf16 v16bf;
typedef __attribute__((ext_vector_type(8)))  __bf16 v8bf;
typedef __attribute__((ext_vector_type(8)))  float  v8f;

union Frag16 {
    v16bf v;
    struct { v8bf lo, hi; } s;
};

// ---------------------------------------------------------------------------
// fp32 -> bf16 conversion of the activations X (8192 x 512)
// ---------------------------------------------------------------------------
__global__ void cvt_x(const float* __restrict__ x, __bf16* __restrict__ xb, int n) {
    int i = blockIdx.x * blockDim.x + threadIdx.x;
    if (i < n) xb[i] = (__bf16)x[i];
}

// ---------------------------------------------------------------------------
// Build padded weight matrix Wb = [trans_W ; state_W ; 0] (NPAD x 512, bf16)
// and concatenated bias vector (NPAD floats).
// ---------------------------------------------------------------------------
__global__ void cvt_w(const float* __restrict__ tw, const float* __restrict__ sw,
                      const float* __restrict__ tb, const float* __restrict__ sb,
                      __bf16* __restrict__ wb, float* __restrict__ bias) {
    int idx = blockIdx.x * blockDim.x + threadIdx.x;  // n * DD + k
    if (idx >= NPAD * DD) return;
    int n = idx / DD;
    int k = idx - n * DD;
    float v = 0.0f;
    if (n < NTRANS)      v = tw[(size_t)n * DD + k];
    else if (n < NTOT)   v = sw[(size_t)(n - NTRANS) * DD + k];
    wb[idx] = (__bf16)v;
    if (k == 0) {
        float bv = 0.0f;
        if (n < NTRANS)    bv = tb[n];
        else if (n < NTOT) bv = sb[n - NTRANS];
        bias[n] = bv;
    }
}

// ---------------------------------------------------------------------------
// Fused potential GEMM: OUT(8192 x NPAD) = X(8192 x 512) * Wb^T + bias
// wave32 WMMA bf16. Each wave computes a 32(M) x 64(N) strip:
//   2 M-tiles x 4 N-tiles = 8 accumulators; B fragments reused across the
//   two M-tiles. A and B fragments are explicitly double-buffered so the
//   loads for k-step n+1 issue before the WMMAs of step n (nonzero
//   s_wait_loadcnt slack instead of wait-0 before every matrix op).
// Block = 256 threads = 8 waves covering 256 M-rows. Grid = (32, 68).
//
// bf16 A-fragment layout (16x32 MxK): lane l holds row m = l&15,
// K = khalf..khalf+7 and khalf+16..khalf+23 with khalf = (l>>4)*8.
// B (32x16 KxN) mirrors it with column n = l&15; since Wb is row-major
// N x K, both fragments are two contiguous 16-byte loads.
// ---------------------------------------------------------------------------
__global__ __launch_bounds__(256)
void gemm_wmma(const __bf16* __restrict__ X, const __bf16* __restrict__ W,
               const float* __restrict__ bias, float* __restrict__ out) {
    const int lane  = threadIdx.x & 31;
    const int wave  = threadIdx.x >> 5;
    const int mlane = lane & 15;
    const int khalf = (lane >> 4) << 3;          // 0 or 8
    const int Mb = blockIdx.x * 256 + wave * 32; // this wave's M base (2 tiles)
    const int Nb = blockIdx.y * 64;              // this block's N base (4 tiles)

    v8f acc[2][4] = {};

    const __bf16* Arow = X + (size_t)(Mb + mlane) * DD + khalf;
    const __bf16* Brow = W + (size_t)(Nb + mlane) * DD + khalf;

    Frag16 a[2][2];   // [buffer][m-tile]
    Frag16 bfr[2][4]; // [buffer][n-tile]

    // prologue: load k-step 0 into buffer 0
#pragma unroll
    for (int mt = 0; mt < 2; ++mt) {
        const __bf16* ap = Arow + (size_t)mt * 16 * DD;
        a[0][mt].s.lo = *(const v8bf*)(ap);
        a[0][mt].s.hi = *(const v8bf*)(ap + 16);
    }
#pragma unroll
    for (int j = 0; j < 4; ++j) {
        const __bf16* bp = Brow + (size_t)j * 16 * DD;
        bfr[0][j].s.lo = *(const v8bf*)(bp);
        bfr[0][j].s.hi = *(const v8bf*)(bp + 16);
    }

#pragma unroll
    for (int ks = 0; ks < DD / 32; ++ks) {
        const int cur = ks & 1;
        const int nxt = cur ^ 1;
        const int k1  = (ks + 1) * 32;
        if (ks + 1 < DD / 32) {
            // issue next k-step's loads before this step's WMMAs
#pragma unroll
            for (int mt = 0; mt < 2; ++mt) {
                const __bf16* ap = Arow + (size_t)mt * 16 * DD + k1;
                a[nxt][mt].s.lo = *(const v8bf*)(ap);
                a[nxt][mt].s.hi = *(const v8bf*)(ap + 16);
            }
#pragma unroll
            for (int j = 0; j < 4; ++j) {
                const __bf16* bp = Brow + (size_t)j * 16 * DD + k1;
                bfr[nxt][j].s.lo = *(const v8bf*)(bp);
                bfr[nxt][j].s.hi = *(const v8bf*)(bp + 16);
            }
            if (k1 + 32 < DD) __builtin_prefetch(Arow + k1 + 32, 0, 1);
        }
#pragma unroll
        for (int j = 0; j < 4; ++j) {
#pragma unroll
            for (int mt = 0; mt < 2; ++mt) {
                acc[mt][j] = __builtin_amdgcn_wmma_f32_16x16x32_bf16(
                    false, a[cur][mt].v, false, bfr[cur][j].v,
                    (short)0, acc[mt][j], false, false);
            }
        }
    }

    // C/D layout: VGPR r, lane l -> row = r + 8*(l>=16), col = l&15
    const int rbase = (lane >> 4) * 8;
#pragma unroll
    for (int j = 0; j < 4; ++j) {
        int col  = Nb + j * 16 + mlane;
        float bv = bias[col];
#pragma unroll
        for (int mt = 0; mt < 2; ++mt) {
#pragma unroll
            for (int r = 0; r < 8; ++r) {
                int row = Mb + mt * 16 + rbase + r;
                out[(size_t)row * NPAD + col] = acc[mt][j][r] + bv;
            }
        }
    }
}

// ---------------------------------------------------------------------------
// Fused partition (logsumexp scan) + Viterbi (max/argmax scan + backtrack).
// One block per batch element; thread j owns label column j.
// sfl[b,t,i,j] = OUT[b*T+t, i*65+j] + OUT[b*T+t, 4225+j]  (biases folded in).
// ---------------------------------------------------------------------------
__global__ __launch_bounds__(128)
void crf_scan(const float* __restrict__ out, float* __restrict__ dst) {
    const int b = blockIdx.x;
    const int j = threadIdx.x;

    __shared__ float lamS[LLAB];
    __shared__ float piS[LLAB - 1];
    __shared__ unsigned char ptrS[TT * (LLAB - 1)]; // 16 KB back-pointer table
    __shared__ int labels[TT];

    const float NEG = -3.0e38f;

    // t = 0: lam = sfl[0, L-1, :], pi0 = sfl[0, L-1, :-1]
    if (j < LLAB) {
        const float* row0 = out + (size_t)b * TT * NPAD; // row m = b*T + 0
        float v = row0[64 * LLAB + j] + row0[NTRANS + j];
        lamS[j] = v;
        if (j < LLAB - 1) piS[j] = v;
    }
    __syncthreads();

    for (int t = 1; t < TT; ++t) {
        float nlam = 0.0f, npi = 0.0f;
        int arg = 0;
        if (j < LLAB) {
            const float* row = out + ((size_t)b * TT + t) * NPAD;
            const float outs = row[NTRANS + j];
            float mx = NEG, s = 0.0f, best = NEG;
            for (int i = 0; i < LLAB; ++i) {
                float v  = row[i * LLAB + j] + outs;
                float vl = v + lamS[i];
                if (vl > mx) { s = s * __expf(mx - vl) + 1.0f; mx = vl; }
                else         { s += __expf(vl - mx); }
                if (i < LLAB - 1 && j < LLAB - 1) {
                    float vp = v + piS[i];
                    if (vp > best) { best = vp; arg = i; }
                }
            }
            nlam = mx + __logf(s);
            npi  = best;
        }
        __syncthreads();
        if (j < LLAB) {
            lamS[j] = nlam;
            if (j < LLAB - 1) {
                piS[j] = npi;
                ptrS[t * (LLAB - 1) + j] = (unsigned char)arg;
            }
        }
        __syncthreads();
    }

    if (j == 0) {
        // Z = lse(lam)
        float mx = NEG;
        for (int i = 0; i < LLAB; ++i) mx = fmaxf(mx, lamS[i]);
        float s = 0.0f;
        for (int i = 0; i < LLAB; ++i) s += __expf(lamS[i] - mx);
        float Z = mx + __logf(s);
        // score / final argmax
        float best = NEG; int bp = 0;
        for (int i = 0; i < LLAB - 1; ++i)
            if (piS[i] > best) { best = piS[i]; bp = i; }
        labels[TT - 1] = bp;
        for (int t = TT - 2; t >= 0; --t) {
            bp = ptrS[(t + 1) * (LLAB - 1) + bp];
            labels[t] = bp;
        }
        float p = __expf(best - Z);
        dst[BB * TT + b] = fminf(fmaxf(p, 0.0f), 1.0f);
    }
    __syncthreads();
    for (int t = j; t < TT; t += blockDim.x)
        dst[b * TT + t] = (float)labels[t];
}

// ---------------------------------------------------------------------------
// Launch: inputs in setup_inputs() order:
//   [0] input_ (B,T,D) f32   [1] state_W (L,D) f32   [2] state_b (L,) f32
//   [3] trans_W (L*L,D) f32  [4] trans_b (L*L,) f32
// d_out: back_pointer (B,T,1) then prob (B,1), flat, as float.
// Workspace layout (~148.3 MB):
//   Xb  bf16 8192*512          =   8,388,608 B
//   Wb  bf16 NPAD*512          =   4,456,448 B
//   bias f32 NPAD              =      17,408 B
//   OUT f32 8192*NPAD          = 142,606,336 B
// ---------------------------------------------------------------------------
extern "C" void kernel_launch(void* const* d_in, const int* in_sizes, int n_in,
                              void* d_out, int out_size, void* d_ws, size_t ws_size,
                              hipStream_t stream) {
    const float* input   = (const float*)d_in[0];
    const float* state_W = (const float*)d_in[1];
    const float* state_b = (const float*)d_in[2];
    const float* trans_W = (const float*)d_in[3];
    const float* trans_b = (const float*)d_in[4];

    char* ws = (char*)d_ws;
    __bf16* Xb  = (__bf16*)ws;
    __bf16* Wb  = (__bf16*)(ws + (size_t)BB * TT * DD * 2);
    float*  bias = (float*)(ws + (size_t)BB * TT * DD * 2 + (size_t)NPAD * DD * 2);
    float*  OUT  = (float*)((char*)bias + (size_t)NPAD * sizeof(float));

    const int nx = BB * TT * DD;
    cvt_x<<<(nx + 255) / 256, 256, 0, stream>>>(input, Xb, nx);

    const int nw = NPAD * DD;
    cvt_w<<<(nw + 255) / 256, 256, 0, stream>>>(trans_W, state_W, trans_b, state_b,
                                                Wb, bias);

    dim3 grid((BB * TT) / 256, NPAD / 64);   // (32, 68)
    gemm_wmma<<<grid, 256, 0, stream>>>(Xb, Wb, bias, OUT);

    crf_scan<<<BB, 128, 0, stream>>>(OUT, (float*)d_out);
}
